// CLIP_64518998720640
// MI455X (gfx1250) — compile-verified, matrix operand order checked by reading, not run on hardware
//
#include <hip/hip_runtime.h>

typedef _Float16 half_t;
typedef _Float16 v16h __attribute__((ext_vector_type(16)));
typedef _Float16 v8h  __attribute__((ext_vector_type(8)));
typedef float    v8f  __attribute__((ext_vector_type(8)));
typedef float    v4f  __attribute__((ext_vector_type(4)));

#define BK 32
#define LDSS 40  // BK + 8 halves padding (keeps 16B alignment, breaks bank phase)

// Model constants
#define B_    16
#define VW_   768
#define VL_   12
#define VH_   12
#define VS_   197
#define NP_   196
#define CTX_  77
#define TW_   512
#define THN_  8
#define TL_   12
#define ED_   512
#define V3_   2304
#define T3_   1536
#define VSP_  224   // 197 padded to mult of 32
#define TSP_  96    // 77 padded to mult of 32
#define VM_   (B_*VS_)
#define TM_   (B_*CTX_)
#define PM_   (B_*NP_)

// ---------------------------------------------------------------------------
// Generic batched NT GEMM:  C[z] = alpha * A[z] * B[z]^T (+bias) (+residual)
// A: f16 [M,K] row-major. B: [N,K] row-major, f16 or fp32 (converted in-flight).
// Batch offset for X = (z/innerDiv)*xOuter + (z%innerDiv)*xInner.
// Output: C16 (f16, bias + optional quickGELU) or C32 (fp32, existing (if beta != 0) + v).
// K must be a multiple of 32 (callers pad). Register-staged, LDS double-buffered,
// one barrier per K-step, GL2 prefetch of the K+2 tile.
// ---------------------------------------------------------------------------
template<int BMv, int BNv, int WR, int WC>
__global__ __launch_bounds__(WR*WC*32) void k_gemm_t(
    int M, int N, int K,
    const half_t* __restrict__ A, int lda, long aOuter, long aInner,
    const void* __restrict__ Bp, int bIsF32, int ldb, long bOuter, long bInner,
    const float* __restrict__ bias,
    half_t* __restrict__ C16, float* __restrict__ C32,
    int ldc, long cOuter, long cInner,
    int innerDiv, float alpha, float beta, int act)
{
  constexpr int THREADS = WR * WC * 32;
  constexpr int WTM = BMv / WR;        // wave tile rows
  constexpr int WTN = BNv / WC;        // wave tile cols
  constexpr int FM  = WTM / 16;
  constexpr int FN  = WTN / 16;
  constexpr int ITERS = (BMv * BK) / (THREADS * 8);  // 8-half chunks per thread

  __shared__ half_t sA[2 * BMv * LDSS];
  __shared__ half_t sB[2 * BNv * LDSS];

  const int tid  = threadIdx.x;
  const int lane = tid & 31;
  const int wave = tid >> 5;
  const int wm   = wave % WR;
  const int wn   = wave / WR;
  const int z    = blockIdx.z;
  const long zo  = z / innerDiv;
  const long zi  = z % innerDiv;
  const half_t* Ab = A + zo * aOuter + zi * aInner;
  const long bOff  = zo * bOuter + zi * bInner;
  const long cOff  = zo * cOuter + zi * cInner;
  const int m0 = blockIdx.x * BMv;
  const int n0 = blockIdx.y * BNv;

  v8f acc[FM][FN];
  const v8f vzero = {0.f,0.f,0.f,0.f,0.f,0.f,0.f,0.f};
  #pragma unroll
  for (int i = 0; i < FM; ++i)
    #pragma unroll
    for (int j = 0; j < FN; ++j) acc[i][j] = vzero;

  // register staging buffers for one K-tile
  v8h aR[ITERS];
  v8h bR[ITERS];
  v4f bF[ITERS][2];

  const v8h hzero = {0,0,0,0,0,0,0,0};
  const v4f fzero = {0.f,0.f,0.f,0.f};

  auto loadStage = [&](int k0) {
    #pragma unroll
    for (int it = 0; it < ITERS; ++it) {
      const int linear = it * (THREADS * 8) + tid * 8;
      const int r = linear >> 5;
      const int c = linear & (BK - 1);
      const int gm = m0 + r;
      aR[it] = hzero;
      if (gm < M) {
        const half_t* as = Ab + (long)gm * lda + (k0 + c);
        aR[it] = *(const v8h*)as;
        if (k0 + 2 * BK < K) __builtin_prefetch((const void*)(as + 2 * BK), 0, 1);
      }
      const int gn = n0 + r;
      bR[it] = hzero;
      bF[it][0] = fzero; bF[it][1] = fzero;
      if (gn < N) {
        if (bIsF32) {
          const float* bs = (const float*)Bp + bOff + (long)gn * ldb + (k0 + c);
          bF[it][0] = *(const v4f*)bs;
          bF[it][1] = *(const v4f*)(bs + 4);
          if (k0 + 2 * BK < K) __builtin_prefetch((const void*)(bs + 2 * BK), 0, 1);
        } else {
          const half_t* bs = (const half_t*)Bp + bOff + (long)gn * ldb + (k0 + c);
          bR[it] = *(const v8h*)bs;
          if (k0 + 2 * BK < K) __builtin_prefetch((const void*)(bs + 2 * BK), 0, 1);
        }
      }
    }
  };

  auto storeStage = [&](int buf) {
    #pragma unroll
    for (int it = 0; it < ITERS; ++it) {
      const int linear = it * (THREADS * 8) + tid * 8;
      const int r = linear >> 5;
      const int c = linear & (BK - 1);
      *(v8h*)&sA[buf * BMv * LDSS + r * LDSS + c] = aR[it];
      if (bIsF32) {
        v8h vb;
        #pragma unroll
        for (int i = 0; i < 4; ++i) {
          vb[i]     = (half_t)bF[it][0][i];
          vb[4 + i] = (half_t)bF[it][1][i];
        }
        *(v8h*)&sB[buf * BNv * LDSS + r * LDSS + c] = vb;
      } else {
        *(v8h*)&sB[buf * BNv * LDSS + r * LDSS + c] = bR[it];
      }
    }
  };

  auto compute = [&](int buf) {
    // fragment layout per ISA: lanes 0-15 hold K[0..7,16..23]; lanes 16-31 K[8..15,24..31]
    const int l  = lane & 15;
    const int hs = (lane >> 4) * 8;
    v16h af[FM], bfv[FN];
    #pragma unroll
    for (int f = 0; f < FM; ++f) {
      const int ra = buf * BMv * LDSS + (wm * WTM + f * 16 + l) * LDSS;
      v8h alo = *(const v8h*)&sA[ra + hs];
      v8h ahi = *(const v8h*)&sA[ra + 16 + hs];
      af[f] = __builtin_shufflevector(alo, ahi, 0,1,2,3,4,5,6,7,8,9,10,11,12,13,14,15);
    }
    #pragma unroll
    for (int f = 0; f < FN; ++f) {
      const int rb = buf * BNv * LDSS + (wn * WTN + f * 16 + l) * LDSS;
      v8h blo = *(const v8h*)&sB[rb + hs];
      v8h bhi = *(const v8h*)&sB[rb + 16 + hs];
      bfv[f] = __builtin_shufflevector(blo, bhi, 0,1,2,3,4,5,6,7,8,9,10,11,12,13,14,15);
    }
    #pragma unroll
    for (int fm = 0; fm < FM; ++fm)
      #pragma unroll
      for (int fn = 0; fn < FN; ++fn)
        acc[fm][fn] = __builtin_amdgcn_wmma_f32_16x16x32_f16(
            false, af[fm], false, bfv[fn], (short)0, acc[fm][fn], false, false);
  };

  const int nk = K / BK;
  loadStage(0);
  storeStage(0);
  __syncthreads();
  for (int ki = 0; ki < nk; ++ki) {
    const int cur = ki & 1;
    const bool hasNext = (ki + 1 < nk);
    if (hasNext) loadStage((ki + 1) * BK);   // global loads fly during compute
    compute(cur);
    if (hasNext) storeStage(1 - cur);        // write the other buffer
    __syncthreads();                          // one barrier per K-step
  }

  // ---- epilogue: C layout = VGPR r -> row r (lanes 0-15) / row 8+r (16-31) ----
  const int rbase = (lane >> 4) * 8;
  const int cl    = lane & 15;
  #pragma unroll
  for (int fm = 0; fm < FM; ++fm) {
    #pragma unroll
    for (int fn = 0; fn < FN; ++fn) {
      const int gn = n0 + wn * WTN + fn * 16 + cl;
      float bv = 0.f;
      if (bias && gn < N) bv = bias[gn];
      #pragma unroll
      for (int r = 0; r < 8; ++r) {
        const int gm = m0 + wm * WTM + fm * 16 + rbase + r;
        if (gm < M && gn < N) {
          float v = acc[fm][fn][r] * alpha + bv;
          const long idx = cOff + (long)gm * ldc + gn;
          if (C16) {
            if (act) v = v / (1.f + __expf(-1.702f * v));   // quickGELU
            C16[idx] = (half_t)v;
          } else {
            float prev = (beta != 0.f) ? C32[idx] : 0.f;
            C32[idx] = prev + v;
          }
        }
      }
    }
  }
}

// ---------------------------------------------------------------------------
// im2col for the 16x16/stride-16 patch conv: out[b*196+p][c*256+i*16+j]
// ---------------------------------------------------------------------------
__global__ __launch_bounds__(256) void k_im2col(const float* __restrict__ img,
                                                half_t* __restrict__ out)
{
  long idx = (long)blockIdx.x * 256 + threadIdx.x;
  const long total = (long)PM_ * VW_;
  if (idx >= total) return;
  int k = (int)(idx % VW_);
  long m = idx / VW_;
  int p = (int)(m % NP_);
  int b = (int)(m / NP_);
  int c = k >> 8;
  int rem = k & 255;
  int i = rem >> 4, j = rem & 15;
  int ph = p / 14, pw = p % 14;
  float v = img[((long)(b * 3 + c) * 224 + (ph * 16 + i)) * 224 + (pw * 16 + j)];
  out[idx] = (half_t)v;
}

// resid[b,s,:] = (s==0 ? cls : conv[b,s-1]) + pos[s]
__global__ __launch_bounds__(256) void k_assemble_vis(const float* __restrict__ conv,
    const float* __restrict__ cls, const float* __restrict__ pos,
    float* __restrict__ resid)
{
  long idx = (long)blockIdx.x * 256 + threadIdx.x;
  const long total = (long)B_ * VS_ * VW_;
  if (idx >= total) return;
  int c = (int)(idx % VW_);
  int s = (int)((idx / VW_) % VS_);
  long b = idx / ((long)VW_ * VS_);
  float v = (s == 0) ? cls[c] : conv[((long)b * NP_ + (s - 1)) * VW_ + c];
  resid[idx] = v + pos[(long)s * VW_ + c];
}

// resid[b,s,:] = tok[text[b,s]] + pos[s]
__global__ __launch_bounds__(256) void k_embed_text(const int* __restrict__ text,
    const float* __restrict__ tok, const float* __restrict__ pos,
    float* __restrict__ resid)
{
  long idx = (long)blockIdx.x * 256 + threadIdx.x;
  const long total = (long)B_ * CTX_ * TW_;
  if (idx >= total) return;
  int c = (int)(idx % TW_);
  int s = (int)((idx / TW_) % CTX_);
  long b = idx / ((long)TW_ * CTX_);
  int t = text[b * CTX_ + s];
  resid[idx] = tok[(long)t * TW_ + c] + pos[(long)s * TW_ + c];
}

// ---------------------------------------------------------------------------
// LayerNorm (fp32 math); optional fp32 and f16 outputs. One block per row.
// ---------------------------------------------------------------------------
__global__ __launch_bounds__(256) void k_layernorm(const float* __restrict__ x, long rsIn,
    const float* __restrict__ g, const float* __restrict__ bb, int D,
    float* __restrict__ o32, long rs32, half_t* __restrict__ o16, long rs16)
{
  const long row = blockIdx.x;
  const float* xr = x + row * rsIn;
  const int tid = threadIdx.x;
  __shared__ float red[256];

  float s = 0.f;
  for (int i = tid; i < D; i += 256) s += xr[i];
  red[tid] = s; __syncthreads();
  for (int o = 128; o > 0; o >>= 1) { if (tid < o) red[tid] += red[tid + o]; __syncthreads(); }
  const float mean = red[0] / (float)D;
  __syncthreads();

  float v = 0.f;
  for (int i = tid; i < D; i += 256) { float d = xr[i] - mean; v += d * d; }
  red[tid] = v; __syncthreads();
  for (int o = 128; o > 0; o >>= 1) { if (tid < o) red[tid] += red[tid + o]; __syncthreads(); }
  const float rstd = rsqrtf(red[0] / (float)D + 1e-5f);

  for (int i = tid; i < D; i += 256) {
    float y = (xr[i] - mean) * rstd * g[i] + bb[i];
    if (o32) o32[row * rs32 + i] = y;
    if (o16) o16[row * rs16 + i] = (half_t)y;
  }
}

// ---------------------------------------------------------------------------
// Row softmax (fp32 in -> f16 probs, zero-padded to Sp cols). causal masks k>q.
// grid.x = Z*S rows; scores row-major [z][S][S]; probs [z][S][Sp].
// ---------------------------------------------------------------------------
__global__ __launch_bounds__(256) void k_softmax(const float* __restrict__ scores,
    int S, int Sp, int causal, half_t* __restrict__ probs)
{
  const long r = blockIdx.x;
  const int q = (int)(r % S);
  const int valid = causal ? (q + 1) : S;
  const float* srow = scores + r * (long)S;
  half_t* prow = probs + r * (long)Sp;
  const int tid = threadIdx.x;
  __shared__ float red[256];

  float mx = -1e30f;
  for (int i = tid; i < valid; i += 256) mx = fmaxf(mx, srow[i]);
  red[tid] = mx; __syncthreads();
  for (int o = 128; o > 0; o >>= 1) { if (tid < o) red[tid] = fmaxf(red[tid], red[tid + o]); __syncthreads(); }
  const float M = red[0];
  __syncthreads();

  float s = 0.f;
  for (int i = tid; i < valid; i += 256) s += __expf(srow[i] - M);
  red[tid] = s; __syncthreads();
  for (int o = 128; o > 0; o >>= 1) { if (tid < o) red[tid] += red[tid + o]; __syncthreads(); }
  const float inv = 1.f / red[0];

  for (int i = tid; i < Sp; i += 256) {
    float v = (i < valid) ? __expf(srow[i] - M) * inv : 0.f;
    prow[i] = (half_t)v;
  }
}

// vt[z][d][s] = s<S ? qkv[b][s][2D + h*64 + d] : 0    (z = b*H + h)
__global__ __launch_bounds__(256) void k_transpose_v(const half_t* __restrict__ qkv,
    int S, int Sp, int D3, int D, int H, half_t* __restrict__ vt)
{
  long idx = (long)blockIdx.x * 256 + threadIdx.x;
  const long total = (long)B_ * H * 64 * Sp;
  if (idx >= total) return;
  int s = (int)(idx % Sp);
  int d = (int)((idx / Sp) % 64);
  long z = idx / ((long)Sp * 64);
  int h = (int)(z % H);
  long b = z / H;
  half_t v = (half_t)0.f;
  if (s < S) v = qkv[((long)b * S + s) * D3 + 2 * D + h * 64 + d];
  vt[idx] = v;
}

__global__ void k_eot(const int* __restrict__ text, int* __restrict__ eot)
{
  int b = threadIdx.x;
  if (b >= B_) return;
  int best = text[b * CTX_]; int bi = 0;
  for (int s = 1; s < CTX_; ++s) {
    int v = text[b * CTX_ + s];
    if (v > best) { best = v; bi = s; }
  }
  eot[b] = bi;
}

// out[b][e] = sum_c in16[row(b)][c] * W[c][e]   (W fp32 [Kd, E] row-major)
__global__ __launch_bounds__(256) void k_proj(const half_t* __restrict__ in16, long rowStride,
    const int* __restrict__ eot, const float* __restrict__ W, int Kd, int E,
    float* __restrict__ out)
{
  int b = blockIdx.x;
  const half_t* row = eot ? (in16 + (long)(b * CTX_ + eot[b]) * rowStride)
                          : (in16 + (long)b * rowStride);
  for (int e = threadIdx.x; e < E; e += 256) {
    float s = 0.f;
    for (int c = 0; c < Kd; ++c) s += (float)row[c] * W[(long)c * E + e];
    out[b * E + e] = s;
  }
}

// Normalize rows, L = exp(scale)*imgf@txtf^T; out[0:256]=L, out[256:512]=L^T
__global__ __launch_bounds__(256) void k_logits(const float* __restrict__ img,
    const float* __restrict__ txt, const float* __restrict__ lscale,
    float* __restrict__ out)
{
  __shared__ float ni[B_], nt[B_];
  const int t = threadIdx.x;
  if (t < B_) {
    float s = 0.f;
    for (int c = 0; c < ED_; ++c) { float v = img[t * ED_ + c]; s += v * v; }
    ni[t] = rsqrtf(s);
  } else if (t < 2 * B_) {
    int b = t - B_;
    float s = 0.f;
    for (int c = 0; c < ED_; ++c) { float v = txt[b * ED_ + c]; s += v * v; }
    nt[b] = rsqrtf(s);
  }
  __syncthreads();
  const float es = __expf(lscale[0]);
  const int i = t >> 4, j = t & 15;
  float s = 0.f;
  for (int c = 0; c < ED_; ++c) s += img[i * ED_ + c] * txt[j * ED_ + c];
  const float L = es * s * ni[i] * nt[j];
  out[i * 16 + j] = L;
  out[256 + j * 16 + i] = L;
}

// ---------------------------------------------------------------------------
extern "C" void kernel_launch(void* const* d_in, const int* in_sizes, int n_in,
                              void* d_out, int out_size, void* d_ws, size_t ws_size,
                              hipStream_t stream)
{
  (void)in_sizes; (void)n_in; (void)out_size; (void)ws_size;

  const float* image      = (const float*)d_in[0];
  const int*   text       = (const int*)  d_in[1];
  const float* v_conv_w   = (const float*)d_in[2];
  const float* v_cls      = (const float*)d_in[3];
  const float* v_pos      = (const float*)d_in[4];
  const float* v_ln_pre_g = (const float*)d_in[5];
  const float* v_ln_pre_b = (const float*)d_in[6];
  const float* v_qkv_w    = (const float*)d_in[7];
  const float* v_qkv_b    = (const float*)d_in[8];
  const float* v_out_w    = (const float*)d_in[9];
  const float* v_out_b    = (const float*)d_in[10];
  const float* v_ln1_g    = (const float*)d_in[11];
  const float* v_ln1_b    = (const float*)d_in[12];
  const float* v_ln2_g    = (const float*)d_in[13];
  const float* v_ln2_b    = (const float*)d_in[14];
  const float* v_fc_w     = (const float*)d_in[15];
  const float* v_fc_b     = (const float*)d_in[16];
  const float* v_pr_w     = (const float*)d_in[17];
  const float* v_pr_b     = (const float*)d_in[18];
  const float* v_ln_post_g= (const float*)d_in[19];
  const float* v_ln_post_b= (const float*)d_in[20];
  const float* v_proj     = (const float*)d_in[21];
  const float* t_tok      = (const float*)d_in[22];
  const float* t_pos      = (const float*)d_in[23];
  const float* t_qkv_w    = (const float*)d_in[24];
  const float* t_qkv_b    = (const float*)d_in[25];
  const float* t_out_w    = (const float*)d_in[26];
  const float* t_out_b    = (const float*)d_in[27];
  const float* t_ln1_g    = (const float*)d_in[28];
  const float* t_ln1_b    = (const float*)d_in[29];
  const float* t_ln2_g    = (const float*)d_in[30];
  const float* t_ln2_b    = (const float*)d_in[31];
  const float* t_fc_w     = (const float*)d_in[32];
  const float* t_fc_b     = (const float*)d_in[33];
  const float* t_pr_w     = (const float*)d_in[34];
  const float* t_pr_b     = (const float*)d_in[35];
  const float* t_lnf_g    = (const float*)d_in[36];
  const float* t_lnf_b    = (const float*)d_in[37];
  const float* t_proj     = (const float*)d_in[38];
  const float* logitscale = (const float*)d_in[39];

  // ---- workspace bump allocator (256B aligned) ----
  size_t off = 0;
  auto alloc = [&](size_t bytes) -> void* {
    void* p = (char*)d_ws + off;
    off += (bytes + 255) & ~(size_t)255;
    return p;
  };
  half_t* aPatch   = (half_t*)alloc((size_t)PM_ * VW_ * 2);
  float*  convOut  = (float*) alloc((size_t)PM_ * VW_ * 4);
  float*  residV   = (float*) alloc((size_t)VM_ * VW_ * 4);
  half_t* xlnV     = (half_t*)alloc((size_t)VM_ * VW_ * 2);
  half_t* qkvV     = (half_t*)alloc((size_t)VM_ * V3_ * 2);
  float*  scoresV  = (float*) alloc((size_t)B_ * VH_ * VS_ * VS_ * 4);
  half_t* probsV   = (half_t*)alloc((size_t)B_ * VH_ * VS_ * VSP_ * 2);
  half_t* vtV      = (half_t*)alloc((size_t)B_ * VH_ * 64 * VSP_ * 2);
  half_t* attnoV   = (half_t*)alloc((size_t)VM_ * VW_ * 2);
  half_t* mlpV     = (half_t*)alloc((size_t)VM_ * 4 * VW_ * 2);
  float*  residT   = (float*) alloc((size_t)TM_ * TW_ * 4);
  half_t* xlnT     = (half_t*)alloc((size_t)TM_ * TW_ * 2);
  half_t* qkvT     = (half_t*)alloc((size_t)TM_ * T3_ * 2);
  float*  scoresT  = (float*) alloc((size_t)B_ * THN_ * CTX_ * CTX_ * 4);
  half_t* probsT   = (half_t*)alloc((size_t)B_ * THN_ * CTX_ * TSP_ * 2);
  half_t* vtT      = (half_t*)alloc((size_t)B_ * THN_ * 64 * TSP_ * 2);
  half_t* attnoT   = (half_t*)alloc((size_t)TM_ * TW_ * 2);
  half_t* mlpT     = (half_t*)alloc((size_t)TM_ * 4 * TW_ * 2);
  half_t* lnpost16 = (half_t*)alloc((size_t)B_ * VW_ * 2);
  half_t* lnf16    = (half_t*)alloc((size_t)TM_ * TW_ * 2);
  int*    eotIdx   = (int*)   alloc((size_t)B_ * 4);
  float*  imgF     = (float*) alloc((size_t)B_ * ED_ * 4);
  float*  txtF     = (float*) alloc((size_t)B_ * ED_ * 4);

  // big tiles (linear layers): 128x128, 8 waves, 8 WMMA/k-step/wave
  auto gemmB = [&](int M, int N, int K,
                   const half_t* A, int lda, long aO, long aI,
                   const void* Bp, int bF32, int ldb, long bO, long bI,
                   const float* bias, half_t* C16, float* C32,
                   int ldc, long cO, long cI, int innerDiv, int Z,
                   float alpha, float beta, int act) {
    dim3 g((M + 127) / 128, (N + 127) / 128, Z);
    k_gemm_t<128,128,4,2><<<g, 256, 0, stream>>>(M, N, K, A, lda, aO, aI, Bp, bF32, ldb, bO, bI,
                                                 bias, C16, C32, ldc, cO, cI, innerDiv, alpha, beta, act);
  };
  // small tiles (attention): 64x64, 4 waves
  auto gemmS = [&](int M, int N, int K,
                   const half_t* A, int lda, long aO, long aI,
                   const void* Bp, int bF32, int ldb, long bO, long bI,
                   const float* bias, half_t* C16, float* C32,
                   int ldc, long cO, long cI, int innerDiv, int Z,
                   float alpha, float beta, int act) {
    dim3 g((M + 63) / 64, (N + 63) / 64, Z);
    k_gemm_t<64,64,2,2><<<g, 128, 0, stream>>>(M, N, K, A, lda, aO, aI, Bp, bF32, ldb, bO, bI,
                                               bias, C16, C32, ldc, cO, cI, innerDiv, alpha, beta, act);
  };

  // ================= vision tower =================
  {
    long tot = (long)PM_ * VW_;
    k_im2col<<<dim3((unsigned)((tot + 255) / 256)), 256, 0, stream>>>(image, aPatch);
  }
  gemmB(PM_, VW_, VW_, aPatch, VW_, 0, 0, v_conv_w, 1, VW_, 0, 0,
        nullptr, nullptr, convOut, VW_, 0, 0, 1, 1, 1.f, 0.f, 0);
  {
    long tot = (long)B_ * VS_ * VW_;
    k_assemble_vis<<<dim3((unsigned)((tot + 255) / 256)), 256, 0, stream>>>(convOut, v_cls, v_pos, residV);
  }
  k_layernorm<<<VM_, 256, 0, stream>>>(residV, VW_, v_ln_pre_g, v_ln_pre_b, VW_,
                                       residV, VW_, nullptr, 0);

  for (int l = 0; l < VL_; ++l) {
    const float* qw  = v_qkv_w + (long)l * V3_ * VW_;
    const float* qb  = v_qkv_b + (long)l * V3_;
    const float* ow  = v_out_w + (long)l * VW_ * VW_;
    const float* ob  = v_out_b + (long)l * VW_;
    const float* fw  = v_fc_w  + (long)l * 4 * VW_ * VW_;
    const float* fb  = v_fc_b  + (long)l * 4 * VW_;
    const float* pw  = v_pr_w  + (long)l * VW_ * 4 * VW_;
    const float* pb  = v_pr_b  + (long)l * VW_;
    const float* l1g = v_ln1_g + (long)l * VW_;
    const float* l1b = v_ln1_b + (long)l * VW_;
    const float* l2g = v_ln2_g + (long)l * VW_;
    const float* l2b = v_ln2_b + (long)l * VW_;

    k_layernorm<<<VM_, 256, 0, stream>>>(residV, VW_, l1g, l1b, VW_, nullptr, 0, xlnV, VW_);
    gemmB(VM_, V3_, VW_, xlnV, VW_, 0, 0, qw, 1, VW_, 0, 0,
          qb, qkvV, nullptr, V3_, 0, 0, 1, 1, 1.f, 0.f, 0);
    // scores = 0.125 * Q K^T  (batched over z = b*H + h)
    gemmS(VS_, VS_, 64, qkvV, V3_, (long)VS_ * V3_, 64,
          (const void*)(qkvV + VW_), 0, V3_, (long)VS_ * V3_, 64,
          nullptr, nullptr, scoresV, VS_, (long)VH_ * VS_ * VS_, (long)VS_ * VS_,
          VH_, B_ * VH_, 0.125f, 0.f, 0);
    k_softmax<<<B_ * VH_ * VS_, 256, 0, stream>>>(scoresV, VS_, VSP_, 0, probsV);
    {
      long tot = (long)B_ * VH_ * 64 * VSP_;
      k_transpose_v<<<dim3((unsigned)((tot + 255) / 256)), 256, 0, stream>>>(
          qkvV, VS_, VSP_, V3_, VW_, VH_, vtV);
    }
    // O = P V
    gemmS(VS_, 64, VSP_, probsV, VSP_, (long)VH_ * VS_ * VSP_, (long)VS_ * VSP_,
          (const void*)vtV, 0, VSP_, (long)VH_ * 64 * VSP_, (long)64 * VSP_,
          nullptr, attnoV, nullptr, VW_, (long)VS_ * VW_, 64,
          VH_, B_ * VH_, 1.f, 0.f, 0);
    // resid += O Wo^T + bo
    gemmB(VM_, VW_, VW_, attnoV, VW_, 0, 0, ow, 1, VW_, 0, 0,
          ob, nullptr, residV, VW_, 0, 0, 1, 1, 1.f, 1.f, 0);
    k_layernorm<<<VM_, 256, 0, stream>>>(residV, VW_, l2g, l2b, VW_, nullptr, 0, xlnV, VW_);
    gemmB(VM_, 4 * VW_, VW_, xlnV, VW_, 0, 0, fw, 1, VW_, 0, 0,
          fb, mlpV, nullptr, 4 * VW_, 0, 0, 1, 1, 1.f, 0.f, 1);
    gemmB(VM_, VW_, 4 * VW_, mlpV, 4 * VW_, 0, 0, pw, 1, 4 * VW_, 0, 0,
          pb, nullptr, residV, VW_, 0, 0, 1, 1, 1.f, 1.f, 0);
  }

  k_layernorm<<<B_, 256, 0, stream>>>(residV, (long)VS_ * VW_, v_ln_post_g, v_ln_post_b, VW_,
                                      nullptr, 0, lnpost16, VW_);
  k_proj<<<B_, 256, 0, stream>>>(lnpost16, VW_, nullptr, v_proj, VW_, ED_, imgF);

  // ================= text tower =================
  {
    long tot = (long)B_ * CTX_ * TW_;
    k_embed_text<<<dim3((unsigned)((tot + 255) / 256)), 256, 0, stream>>>(text, t_tok, t_pos, residT);
  }
  for (int l = 0; l < TL_; ++l) {
    const float* qw  = t_qkv_w + (long)l * T3_ * TW_;
    const float* qb  = t_qkv_b + (long)l * T3_;
    const float* ow  = t_out_w + (long)l * TW_ * TW_;
    const float* ob  = t_out_b + (long)l * TW_;
    const float* fw  = t_fc_w  + (long)l * 4 * TW_ * TW_;
    const float* fb  = t_fc_b  + (long)l * 4 * TW_;
    const float* pw  = t_pr_w  + (long)l * TW_ * 4 * TW_;
    const float* pb  = t_pr_b  + (long)l * TW_;
    const float* l1g = t_ln1_g + (long)l * TW_;
    const float* l1b = t_ln1_b + (long)l * TW_;
    const float* l2g = t_ln2_g + (long)l * TW_;
    const float* l2b = t_ln2_b + (long)l * TW_;

    k_layernorm<<<TM_, 256, 0, stream>>>(residT, TW_, l1g, l1b, TW_, nullptr, 0, xlnT, TW_);
    gemmB(TM_, T3_, TW_, xlnT, TW_, 0, 0, qw, 1, TW_, 0, 0,
          qb, qkvT, nullptr, T3_, 0, 0, 1, 1, 1.f, 0.f, 0);
    gemmS(CTX_, CTX_, 64, qkvT, T3_, (long)CTX_ * T3_, 64,
          (const void*)(qkvT + TW_), 0, T3_, (long)CTX_ * T3_, 64,
          nullptr, nullptr, scoresT, CTX_, (long)THN_ * CTX_ * CTX_, (long)CTX_ * CTX_,
          THN_, B_ * THN_, 0.125f, 0.f, 0);
    k_softmax<<<B_ * THN_ * CTX_, 256, 0, stream>>>(scoresT, CTX_, TSP_, 1, probsT);
    {
      long tot = (long)B_ * THN_ * 64 * TSP_;
      k_transpose_v<<<dim3((unsigned)((tot + 255) / 256)), 256, 0, stream>>>(
          qkvT, CTX_, TSP_, T3_, TW_, THN_, vtT);
    }
    gemmS(CTX_, 64, TSP_, probsT, TSP_, (long)THN_ * CTX_ * TSP_, (long)CTX_ * TSP_,
          (const void*)vtT, 0, TSP_, (long)THN_ * 64 * TSP_, (long)64 * TSP_,
          nullptr, attnoT, nullptr, TW_, (long)CTX_ * TW_, 64,
          THN_, B_ * THN_, 1.f, 0.f, 0);
    gemmB(TM_, TW_, TW_, attnoT, TW_, 0, 0, ow, 1, TW_, 0, 0,
          ob, nullptr, residT, TW_, 0, 0, 1, 1, 1.f, 1.f, 0);
    k_layernorm<<<TM_, 256, 0, stream>>>(residT, TW_, l2g, l2b, TW_, nullptr, 0, xlnT, TW_);
    gemmB(TM_, 4 * TW_, TW_, xlnT, TW_, 0, 0, fw, 1, TW_, 0, 0,
          fb, mlpT, nullptr, 4 * TW_, 0, 0, 1, 1, 1.f, 0.f, 1);
    gemmB(TM_, TW_, 4 * TW_, mlpT, 4 * TW_, 0, 0, pw, 1, 4 * TW_, 0, 0,
          pb, nullptr, residT, TW_, 0, 0, 1, 1, 1.f, 1.f, 0);
  }

  k_layernorm<<<TM_, 256, 0, stream>>>(residT, TW_, t_lnf_g, t_lnf_b, TW_,
                                       nullptr, 0, lnf16, TW_);
  k_eot<<<1, 32, 0, stream>>>(text, eotIdx);
  k_proj<<<B_, 256, 0, stream>>>(lnf16, TW_, eotIdx, t_proj, TW_, ED_, txtF);

  // ================= similarity =================
  k_logits<<<1, 256, 0, stream>>>(imgF, txtF, logitscale, (float*)d_out);
}
